// Attention_46462956208682
// MI455X (gfx1250) — matrix operand order repaired and checked by reading, not compiled, not run on hardware
//
#include <hip/hip_runtime.h>

// ---------------- problem constants ----------------
#define BB   2
#define TT   2048
#define CC   2048
#define HH   16
#define HKVN 4
#define DD   128
#define GROUP (HH / HKVN)

typedef __attribute__((ext_vector_type(16))) __bf16 bf16x16;
typedef __attribute__((ext_vector_type(8)))  __bf16 bf16x8;
typedef __attribute__((ext_vector_type(4)))  __bf16 bf16x4;
typedef __attribute__((ext_vector_type(8)))  float  f32x8;

// ---------------- bf16 helpers ----------------
__device__ __forceinline__ __bf16 f2bf(float f) {
    unsigned u = __builtin_bit_cast(unsigned, f);
    unsigned r = u + 0x7FFFu + ((u >> 16) & 1u);   // round-to-nearest-even
    unsigned short h = (unsigned short)(r >> 16);
    return __builtin_bit_cast(__bf16, h);
}
__device__ __forceinline__ float bf2f(__bf16 b) {
    unsigned short h = __builtin_bit_cast(unsigned short, b);
    unsigned u = ((unsigned)h) << 16;
    return __builtin_bit_cast(float, u);
}

// ---------------- CDNA5 async global->LDS copy (16 bytes / lane) ----------------
// GLOBAL_LOAD_ASYNC_TO_LDS_B128 (ISA 15.18.3 op 98): VDST = LDS byte address,
// VADDR = 64-bit global address. Tracked by ASYNCcnt. LDS offset = low 32 bits
// of the generic pointer (aperture keeps AS3 offset in addr[31:0], ISA 10.2).
__device__ __forceinline__ void async_copy16(const __bf16* gsrc, __bf16* ldst) {
    unsigned lds = (unsigned)(size_t)(void*)ldst;
    asm volatile("global_load_async_to_lds_b128 %0, %1, off"
                 :: "v"(lds), "v"(gsrc) : "memory");
}
__device__ __forceinline__ void wait_async() {
    asm volatile("s_wait_asynccnt 0x0" ::: "memory");
}

// ---------------- WMMA fragment loaders (ISA 7.12.2 layouts) ----------------
// A 16x32 bf16: lane holds row M=lane%16; halves 0..7 -> K = k0+(lane>=16?8:0)+h,
// halves 8..15 -> K = k0+16+(lane>=16?8:0)+(h-8). Two contiguous 16B runs.
__device__ __forceinline__ bf16x16 load_a_frag(const __bf16* base, int stride,
                                               int row0, int k0) {
    int lane = threadIdx.x & 31;
    const __bf16* p = base + (size_t)(row0 + (lane & 15)) * stride
                           + k0 + ((lane >> 4) << 3);
    bf16x8 lo = *(const bf16x8*)(p);
    bf16x8 hi = *(const bf16x8*)(p + 16);
    bf16x16 a;
#pragma unroll
    for (int i = 0; i < 8; ++i) { a[i] = lo[i]; a[i + 8] = hi[i]; }
    return a;
}
// B 32x16 bf16: lane holds col N=lane%16; halves 0..7 -> K = k0+(lane>=16?8:0)+h,
// halves 8..15 -> K = k0+16+(lane>=16?8:0)+(h-8). Two contiguous 16B runs.
__device__ __forceinline__ bf16x16 load_b_frag(const __bf16* base, int stride,
                                               int col0, int k0) {
    int lane = threadIdx.x & 31;
    const __bf16* p = base + (size_t)(col0 + (lane & 15)) * stride
                           + k0 + ((lane >> 4) << 3);
    bf16x8 lo = *(const bf16x8*)(p);
    bf16x8 hi = *(const bf16x8*)(p + 16);
    bf16x16 b;
#pragma unroll
    for (int i = 0; i < 8; ++i) { b[i] = lo[i]; b[i + 8] = hi[i]; }
    return b;
}
__device__ __forceinline__ f32x8 wmma_bf16(bf16x16 a, bf16x16 b, f32x8 c) {
    return __builtin_amdgcn_wmma_f32_16x16x32_bf16(
        false, a, false, b, (short)0, c, false, false);
}

// ---------------- kernel: f32 -> bf16 convert (4 elems/thread) ----------------
__global__ void cvt_kernel(const float* __restrict__ in, __bf16* __restrict__ out,
                           int n4) {
    int idx = blockIdx.x * blockDim.x + threadIdx.x;
    int stride = gridDim.x * blockDim.x;
    for (int i = idx; i < n4; i += stride) {
        float4 f = ((const float4*)in)[i];
        bf16x4 o;
        o[0] = f2bf(f.x); o[1] = f2bf(f.y); o[2] = f2bf(f.z); o[3] = f2bf(f.w);
        ((bf16x4*)out)[i] = o;
    }
}

// ---------------- kernel: in-place RoPE (interleaved pairs) ----------------
// x layout (b, h, t, d); pair (2i, 2i+1); optional scale folded in (q: 1/sqrt(D)).
__global__ void rope_kernel(__bf16* __restrict__ x, float scale, int total) {
    int idx = blockIdx.x * blockDim.x + threadIdx.x;
    if (idx >= total) return;
    int i  = idx & (DD / 2 - 1);          // 64 pairs
    int t  = (idx >> 6) & (TT - 1);
    int bh = idx >> 17;                   // / (64 * 2048)
    __bf16* p = x + ((size_t)bh * TT + t) * DD + 2 * i;
    float x1 = bf2f(p[0]), x2 = bf2f(p[1]);
    float ex   = (float)(2 * i) / (float)DD;
    float freq = __expf(-ex * 13.815510557964274f);   // ln(1e6)
    float c, s;
    __sincosf((float)t * freq, &c, &s);
    p[0] = f2bf((x1 * c - x2 * s) * scale);
    p[1] = f2bf((x1 * s + x2 * c) * scale);
}

// ---------------- kernel: bf16 WMMA GEMM  C[m,n] = sum_k A[m,k] * W[n,k] -------
// Double-buffered W tile staged via async global->LDS; A fragments from global.
// mode 0: write bf16 to (b, h, t, d) layout (Hx heads).  mode 1: f32 row-major.
#define LDW_S 48     // 64x32 tile, 96B rows (16B-aligned chunks, de-phased banks)
__global__ __launch_bounds__(128)
void gemm_bf16_kernel(const __bf16* __restrict__ A, const __bf16* __restrict__ W,
                      int M, int N, int K,
                      __bf16* __restrict__ obf, float* __restrict__ of32,
                      int mode, int Hx) {
    __shared__ __bf16 lds_w[2][64 * LDW_S];
    const int tid  = threadIdx.x;
    const int wave = tid >> 5;
    const int lane = tid & 31;
    const int n0   = blockIdx.x * 64;
    const int m0   = blockIdx.y * 64 + wave * 16;

    // each thread stages two 16B chunks per tile: c = tid and tid+128 (256 total)
    const int r0 = tid >> 2,          c0 = (tid & 3) << 3;
    const int r1 = (tid + 128) >> 2,  c1 = ((tid + 128) & 3) << 3;
    const __bf16* w0 = W + (size_t)(n0 + r0) * K + c0;
    const __bf16* w1 = W + (size_t)(n0 + r1) * K + c1;

    f32x8 acc[4];
#pragma unroll
    for (int j = 0; j < 4; ++j) acc[j] = {};

    const int nk = K >> 5;
    // prologue: stage tile 0 into buffer 0
    async_copy16(w0, lds_w[0] + r0 * LDW_S + c0);
    async_copy16(w1, lds_w[0] + r1 * LDW_S + c1);

    for (int i = 0; i < nk; ++i) {
        wait_async();          // own tile-i copies (and older) have landed
        __syncthreads();       // everyone's copies landed; prev compute done
        if (i + 1 < nk) {      // prefetch next tile into the other buffer
            int kk = (i + 1) << 5;
            __bf16* dst = lds_w[(i + 1) & 1];
            async_copy16(w0 + kk, dst + r0 * LDW_S + c0);
            async_copy16(w1 + kk, dst + r1 * LDW_S + c1);
        }
        const __bf16* wb = lds_w[i & 1];
        bf16x16 a = load_a_frag(A + (size_t)m0 * K, K, 0, i << 5);
#pragma unroll
        for (int j = 0; j < 4; ++j) {
            bf16x16 bw = load_b_frag(wb, LDW_S, j * 16, 0);
            acc[j] = wmma_bf16(a, bw, acc[j]);
        }
    }

    const int roff = (lane >> 4) << 3;
#pragma unroll
    for (int j = 0; j < 4; ++j) {
#pragma unroll
        for (int r = 0; r < 8; ++r) {
            int m = m0 + roff + r;
            int n = n0 + j * 16 + (lane & 15);
            float v = acc[j][r];
            if (mode == 0) {
                int bidx = m >> 11, t = m & (TT - 1);
                int hh   = n >> 7,  d = n & (DD - 1);
                obf[(((size_t)bidx * Hx + hh) * TT + t) * DD + d] = f2bf(v);
            } else {
                of32[(size_t)m * N + n] = v;
            }
        }
    }
}

// ---------------- kernel: causal flash attention (GQA) ----------------
// grid (T/64, H, B), 128 threads (4 waves); wave w owns Q rows [qt*64+16w, +16).
#define LDK_S 144   // 64 x 128 K tile rows, padded (288B rows, 16/32B aligned)
#define LDV_S 80    // 128 x 64 transposed V tile (160B rows)
#define LDP_S 80    // per-wave 16 x 64 P tile
__global__ __launch_bounds__(128)
void flash_kernel(const __bf16* __restrict__ Q, const __bf16* __restrict__ Kb,
                  const __bf16* __restrict__ Vb, __bf16* __restrict__ O) {
    __shared__ __bf16 lds_k [64  * LDK_S];
    __shared__ __bf16 lds_vt[128 * LDV_S];
    __shared__ __bf16 lds_p [4 * 16 * LDP_S];

    const int tid  = threadIdx.x;
    const int wave = tid >> 5;
    const int lane = tid & 31;
    const int qt   = blockIdx.x;
    const int h    = blockIdx.y;
    const int b    = blockIdx.z;
    const int hkv  = h / GROUP;
    const int roff = (lane >> 4) << 3;

    // Q fragments for this wave's 16 rows, all of D=128 (4 k-steps), from global
    const __bf16* qg = Q + (((size_t)b * HH + h) * TT + qt * 64 + wave * 16) * DD;
    bf16x16 qf[4];
#pragma unroll
    for (int kk = 0; kk < 4; ++kk) qf[kk] = load_a_frag(qg, DD, 0, kk * 32);

    f32x8 oacc[8];
#pragma unroll
    for (int d = 0; d < 8; ++d) oacc[d] = {};
    float mrun[8], lrun[8];
#pragma unroll
    for (int r = 0; r < 8; ++r) { mrun[r] = -1e30f; lrun[r] = 0.f; }

    __bf16* pbase = lds_p + wave * 16 * LDP_S;

    for (int kt = 0; kt <= qt; ++kt) {
        __syncthreads();   // all waves done reading previous K/V tile
        const __bf16* kg = Kb + (((size_t)b * HKVN + hkv) * TT + kt * 64) * DD;
        const __bf16* vg = Vb + (((size_t)b * HKVN + hkv) * TT + kt * 64) * DD;
        for (int c = tid; c < 1024; c += 128) {       // 64x128 / 8
            int row = c >> 4;
            int col = (c & 15) << 3;
            // K tile: async DMA straight into LDS (no VGPR round-trip)
            async_copy16(kg + row * DD + col, lds_k + row * LDK_S + col);
            // V tile: transpose on the way in (needs VGPR staging)
            bf16x8 vv = *(const bf16x8*)(vg + row * DD + col);
#pragma unroll
            for (int i = 0; i < 8; ++i) lds_vt[(col + i) * LDV_S + row] = vv[i];
        }
        wait_async();
        __syncthreads();

        // S = Q K^T : 16 x 64 (4 n-subtiles x 4 k-steps)
        f32x8 s[4];
#pragma unroll
        for (int j = 0; j < 4; ++j) {
            f32x8 sa = {};
#pragma unroll
            for (int kk = 0; kk < 4; ++kk) {
                bf16x16 bk = load_b_frag(lds_k, LDK_S, j * 16, kk * 32);
                sa = wmma_bf16(qf[kk], bk, sa);
            }
            s[j] = sa;
        }

        // online softmax (scale already folded into Q)
        const bool diag = (kt == qt);
        float rmax[8];
#pragma unroll
        for (int r = 0; r < 8; ++r) rmax[r] = -1e30f;
#pragma unroll
        for (int j = 0; j < 4; ++j) {
#pragma unroll
            for (int r = 0; r < 8; ++r) {
                float v = s[j][r];
                if (diag) {
                    int trow = wave * 16 + roff + r;
                    int scol = j * 16 + (lane & 15);
                    if (scol > trow) v = -1e9f;
                }
                s[j][r] = v;
                rmax[r] = fmaxf(rmax[r], v);
            }
        }
#pragma unroll
        for (int r = 0; r < 8; ++r) {
#pragma unroll
            for (int m = 1; m < 16; m <<= 1)   // stays inside each 16-lane half
                rmax[r] = fmaxf(rmax[r], __shfl_xor(rmax[r], m, 32));
        }
        float alpha[8];
#pragma unroll
        for (int r = 0; r < 8; ++r) {
            float mn = fmaxf(mrun[r], rmax[r]);
            alpha[r] = __expf(mrun[r] - mn);
            mrun[r]  = mn;
        }
        float rsum[8];
#pragma unroll
        for (int r = 0; r < 8; ++r) rsum[r] = 0.f;
#pragma unroll
        for (int j = 0; j < 4; ++j) {
#pragma unroll
            for (int r = 0; r < 8; ++r) {
                float e = __expf(s[j][r] - mrun[r]);
                rsum[r] += e;
                pbase[(roff + r) * LDP_S + j * 16 + (lane & 15)] = f2bf(e);
            }
        }
#pragma unroll
        for (int r = 0; r < 8; ++r) {
#pragma unroll
            for (int m = 1; m < 16; m <<= 1)
                rsum[r] += __shfl_xor(rsum[r], m, 32);
            lrun[r] = lrun[r] * alpha[r] + rsum[r];
        }
#pragma unroll
        for (int d = 0; d < 8; ++d)
#pragma unroll
            for (int r = 0; r < 8; ++r)
                oacc[d][r] *= alpha[r];

        // O += P V  (P from per-wave LDS; LDS ops are in-order per wave)
        bf16x16 p0 = load_a_frag(pbase, LDP_S, 0, 0);
        bf16x16 p1 = load_a_frag(pbase, LDP_S, 0, 32);
#pragma unroll
        for (int d = 0; d < 8; ++d) {
            bf16x16 b0 = load_b_frag(lds_vt, LDV_S, d * 16, 0);
            bf16x16 b1 = load_b_frag(lds_vt, LDV_S, d * 16, 32);
            oacc[d] = wmma_bf16(p0, b0, oacc[d]);
            oacc[d] = wmma_bf16(p1, b1, oacc[d]);
        }
    }

    // write O in (b, t, h*D) layout
#pragma unroll
    for (int d = 0; d < 8; ++d) {
#pragma unroll
        for (int r = 0; r < 8; ++r) {
            int t  = qt * 64 + wave * 16 + roff + r;
            int dd = d * 16 + (lane & 15);
            float v = oacc[d][r] / lrun[r];
            O[((size_t)b * TT + t) * (HH * DD) + h * DD + dd] = f2bf(v);
        }
    }
}

// ---------------- host launch ----------------
extern "C" void kernel_launch(void* const* d_in, const int* in_sizes, int n_in,
                              void* d_out, int out_size, void* d_ws, size_t ws_size,
                              hipStream_t stream) {
    const float* x  = (const float*)d_in[0];
    // d_in[1] = mask (causal mask applied analytically in flash kernel)
    const float* wq = (const float*)d_in[2];
    const float* wk = (const float*)d_in[3];
    const float* wv = (const float*)d_in[4];
    const float* wo = (const float*)d_in[5];
    float* out = (float*)d_out;

    const size_t n_x  = (size_t)BB * TT * CC;        // 8,388,608
    const size_t n_wq = (size_t)HH * DD * CC;        // 4,194,304
    const size_t n_wk = (size_t)HKVN * DD * CC;      // 1,048,576
    const size_t n_q  = (size_t)BB * HH * TT * DD;   // 8,388,608
    const size_t n_k  = (size_t)BB * HKVN * TT * DD; // 2,097,152

    __bf16* p   = (__bf16*)d_ws;
    __bf16* xb  = p; p += n_x;
    __bf16* wqb = p; p += n_wq;
    __bf16* wkb = p; p += n_wk;
    __bf16* wvb = p; p += n_wk;
    __bf16* wob = p; p += n_wq;
    __bf16* Qb  = p; p += n_q;
    __bf16* Kb  = p; p += n_k;
    __bf16* Vb  = p; p += n_k;
    __bf16* Ob  = p; p += n_x;   // ~80 MB total workspace

    cvt_kernel<<<2048, 256, 0, stream>>>(x,  xb,  (int)(n_x  / 4));
    cvt_kernel<<<1024, 256, 0, stream>>>(wq, wqb, (int)(n_wq / 4));
    cvt_kernel<<<512,  256, 0, stream>>>(wk, wkb, (int)(n_wk / 4));
    cvt_kernel<<<512,  256, 0, stream>>>(wv, wvb, (int)(n_wk / 4));
    cvt_kernel<<<1024, 256, 0, stream>>>(wo, wob, (int)(n_wq / 4));

    const int M = BB * TT;   // 4096
    gemm_bf16_kernel<<<dim3((HH * DD) / 64,  M / 64), 128, 0, stream>>>(
        xb, wqb, M, HH * DD,  CC, Qb, nullptr, 0, HH);
    gemm_bf16_kernel<<<dim3((HKVN * DD) / 64, M / 64), 128, 0, stream>>>(
        xb, wkb, M, HKVN * DD, CC, Kb, nullptr, 0, HKVN);
    gemm_bf16_kernel<<<dim3((HKVN * DD) / 64, M / 64), 128, 0, stream>>>(
        xb, wvb, M, HKVN * DD, CC, Vb, nullptr, 0, HKVN);

    const float scale = 0.08838834764831845f;   // 1/sqrt(128), folded into Q
    int totq = (int)(BB * HH   * TT * (DD / 2));
    int totk = (int)(BB * HKVN * TT * (DD / 2));
    rope_kernel<<<(totq + 255) / 256, 256, 0, stream>>>(Qb, scale, totq);
    rope_kernel<<<(totk + 255) / 256, 256, 0, stream>>>(Kb, 1.0f,  totk);

    flash_kernel<<<dim3(TT / 64, HH, BB), 128, 0, stream>>>(Qb, Kb, Vb, Ob);

    gemm_bf16_kernel<<<dim3(CC / 64, M / 64), 128, 0, stream>>>(
        Ob, wob, M, CC, HH * DD, nullptr, out, 1, 0);
}
